// ACARHead_4552665334426
// MI455X (gfx1250) — matrix operand order repaired and checked by reading, not compiled, not run on hardware
//
#include <hip/hip_runtime.h>
#include <hip/hip_bf16.h>

// ---------------------------------------------------------------------------
// CDNA5 (gfx1250) WMMA building blocks: wave32, v_wmma_f32_16x16x32_bf16
// ---------------------------------------------------------------------------
typedef __attribute__((ext_vector_type(16))) __bf16 v16bf;
typedef __attribute__((ext_vector_type(8)))  __bf16 v8bf;
typedef __attribute__((ext_vector_type(4)))  __bf16 v4bf;
typedef __attribute__((ext_vector_type(8)))  float  v8f;
typedef __attribute__((ext_vector_type(4)))  float  v4f;

#define WMMA_BF16(a_, b_, c_) \
  __builtin_amdgcn_wmma_f32_16x16x32_bf16(false, (a_), false, (b_), (short)0, (c_), false, false)

// A fragment (16x32, bf16): lane l<16 -> M=l, elems i = K[g*16 + i];
// g = lane/16. Two 16B LDS loads per lane.
__device__ __forceinline__ v16bf load_a_frag(const __bf16* sA, int ldk, int mrow,
                                             int kbase, int lane) {
  int l = lane & 15, g = (lane >> 4) & 1;
  const __bf16* base = sA + (mrow + l) * ldk + kbase;
  v8bf lo = *(const v8bf*)(base + g * 16);
  v8bf hi = *(const v8bf*)(base + g * 16 + 8);
  v16bf a;
#pragma unroll
  for (int i = 0; i < 8; ++i) { a[i] = lo[i]; a[8 + i] = hi[i]; }
  return a;
}

// B fragment (32x16, bf16), LDS holds B transposed [N][K] so K is contiguous:
// lane l<16 -> N=l, elems i = K[g*16 + i]. One 32B LDS load per lane.
__device__ __forceinline__ v16bf load_b_frag(const __bf16* sBt, int ldk, int ncol,
                                             int kbase, int lane) {
  int l = lane & 15, g = (lane >> 4) & 1;
  return *(const v16bf*)(sBt + (ncol + l) * ldk + kbase + g * 16);
}

// A fragment gathered from an f32 LDS buffer (attention matrix), cvt to bf16.
__device__ __forceinline__ v16bf load_a_frag_f32(const float* sAf, int ldk, int mrow,
                                                 int kbase, int lane) {
  int l = lane & 15, g = (lane >> 4) & 1;
  const float* base = sAf + (mrow + l) * ldk + kbase;
  v16bf a;
#pragma unroll
  for (int i = 0; i < 8; ++i) {
    a[i]     = (__bf16)base[g * 16 + i];
    a[8 + i] = (__bf16)base[g * 16 + 8 + i];
  }
  return a;
}

// Stage a 64x64 fp32->bf16 tile, row-major in LDS (weights / A matrix).
// Load phase first (4 independent b128 loads), then convert+store phase, so
// the scheduler batches VMEM and uses descending s_wait_loadcnt.
__device__ __forceinline__ void stage_a64x64(const float* __restrict__ A, int lda,
                                             int mBase, int k0, int M, int K,
                                             __bf16* sA, int tid) {
  v4f f[4];
  bool oks[4];
#pragma unroll
  for (int t = 0; t < 4; ++t) {
    int v  = tid + t * 256;           // 0..1023
    int r  = v >> 4;                  // 0..63
    int c4 = (v & 15) << 2;           // 0..60
    int gm = mBase + r, gk = k0 + c4;
    bool ok = (gm < M) & ((gk + 3) < K);
    oks[t] = ok;
    f[t] = *(const v4f*)(A + (ok ? ((long)gm * lda + gk) : 0));
    if (t == 0) __builtin_prefetch(A + (long)gm * lda + gk + 64, 0, 1);
  }
#pragma unroll
  for (int t = 0; t < 4; ++t) {
    int v  = tid + t * 256;
    int r  = v >> 4;
    int c4 = (v & 15) << 2;
    v4bf pk;
#pragma unroll
    for (int j = 0; j < 4; ++j) pk[j] = (__bf16)(oks[t] ? f[t][j] : 0.f);
    *(v4bf*)&sA[r * 64 + c4] = pk;
  }
}

// ---------------------------------------------------------------------------
// Generic tiled GEMM: C[M,N] = op(A[M,K] * B) (+relu). fp32 in/out, bf16 WMMA.
// Block tile 64x128xBK64, 256 threads = 8 waves (2x4), wave = 32x32 (2x2 frags).
// bNK==1: B is [N,K] row-major. bNK==0: B is [K,N] row-major.
// ---------------------------------------------------------------------------
__global__ __launch_bounds__(256) void gemm_bf16_wmma(
    const float* __restrict__ A, int lda,
    const float* __restrict__ B, int ldb, int bNK,
    float* __restrict__ C, int ldc,
    int M, int N, int K, int relu) {
  __shared__ alignas(32) __bf16 sA[64 * 64];
  __shared__ alignas(32) __bf16 sBt[128 * 64];
  int tid = threadIdx.x, lane = tid & 31, w = tid >> 5;
  int wm = w >> 2, wn = w & 3;
  int mBase = blockIdx.y * 64, nBase = blockIdx.x * 128;
  v8f acc[2][2] = {};
  for (int k0 = 0; k0 < K; k0 += 64) {
    stage_a64x64(A, lda, mBase, k0, M, K, sA, tid);
    if (bNK) {
      // B[N,K]: contiguous along K -> vectorized; load phase then store phase
      v4f f[8];
      bool oks[8];
#pragma unroll
      for (int t = 0; t < 8; ++t) {
        int v  = tid + t * 256;       // 0..2047
        int r  = v >> 4;              // 0..127
        int c4 = (v & 15) << 2;
        int gn = nBase + r, gk = k0 + c4;
        bool ok = (gn < N) & ((gk + 3) < K);
        oks[t] = ok;
        f[t] = *(const v4f*)(B + (ok ? ((long)gn * ldb + gk) : 0));
      }
#pragma unroll
      for (int t = 0; t < 8; ++t) {
        int v  = tid + t * 256;
        int r  = v >> 4;
        int c4 = (v & 15) << 2;
        v4bf pk;
#pragma unroll
        for (int j = 0; j < 4; ++j) pk[j] = (__bf16)(oks[t] ? f[t][j] : 0.f);
        *(v4bf*)&sBt[r * 64 + c4] = pk;
      }
    } else {
      // B[K,N]: contiguous along N -> vector load, transposed LDS stores
      v4f f[8];
      bool oks[8];
#pragma unroll
      for (int t = 0; t < 8; ++t) {
        int v  = tid + t * 256;       // 0..2047
        int kk = v >> 5;              // 0..63
        int n4 = (v & 31) << 2;       // 0..124
        int gk = k0 + kk, gn = nBase + n4;
        bool ok = (gk < K) & ((gn + 3) < N);
        oks[t] = ok;
        f[t] = *(const v4f*)(B + (ok ? ((long)gk * ldb + gn) : 0));
      }
#pragma unroll
      for (int t = 0; t < 8; ++t) {
        int v  = tid + t * 256;
        int kk = v >> 5;
        int n4 = (v & 31) << 2;
#pragma unroll
        for (int j = 0; j < 4; ++j)
          sBt[(n4 + j) * 64 + kk] = (__bf16)(oks[t] ? f[t][j] : 0.f);
      }
    }
    __syncthreads();
    v16bf a00 = load_a_frag(sA, 64, wm * 32,      0,  lane);
    v16bf a01 = load_a_frag(sA, 64, wm * 32,      32, lane);
    v16bf a10 = load_a_frag(sA, 64, wm * 32 + 16, 0,  lane);
    v16bf a11 = load_a_frag(sA, 64, wm * 32 + 16, 32, lane);
    v16bf b00 = load_b_frag(sBt, 64, wn * 32,      0,  lane);
    v16bf b01 = load_b_frag(sBt, 64, wn * 32,      32, lane);
    v16bf b10 = load_b_frag(sBt, 64, wn * 32 + 16, 0,  lane);
    v16bf b11 = load_b_frag(sBt, 64, wn * 32 + 16, 32, lane);
    // same-acc WMMAs kept 4 apart (bf16 WMMA->WMMA RAW hazard distance)
    acc[0][0] = WMMA_BF16(a00, b00, acc[0][0]);
    acc[0][1] = WMMA_BF16(a00, b10, acc[0][1]);
    acc[1][0] = WMMA_BF16(a10, b00, acc[1][0]);
    acc[1][1] = WMMA_BF16(a10, b10, acc[1][1]);
    acc[0][0] = WMMA_BF16(a01, b01, acc[0][0]);
    acc[0][1] = WMMA_BF16(a01, b11, acc[0][1]);
    acc[1][0] = WMMA_BF16(a11, b01, acc[1][0]);
    acc[1][1] = WMMA_BF16(a11, b11, acc[1][1]);
    __syncthreads();
  }
  int l = lane & 15, g = (lane >> 4) & 1;
#pragma unroll
  for (int mi = 0; mi < 2; ++mi)
#pragma unroll
    for (int ni = 0; ni < 2; ++ni)
#pragma unroll
      for (int r = 0; r < 8; ++r) {
        int row = mBase + wm * 32 + mi * 16 + g * 8 + r;
        int col = nBase + wn * 32 + ni * 16 + l;
        if (row < M && col < N) {
          float v = acc[mi][ni][r];
          if (relu) v = fmaxf(v, 0.f);
          C[row * ldc + col] = v;
        }
      }
}

// ---------------------------------------------------------------------------
// Implicit-GEMM 3x3 conv: out[n,o,oy,ox] = sum_{c,kh,kw} w[o,c,kh,kw]*in[n,c..]
// M = O (weights, direct rows), N = Nb*OH*OW (pixels), K = C*9 (im2col gather
// in batches of 8 independent branchless loads). Fused residual / ReLU.
// ---------------------------------------------------------------------------
__global__ __launch_bounds__(256) void conv3x3_wmma(
    const float* __restrict__ in, const float* __restrict__ wgt,
    float* __restrict__ out, const float* __restrict__ res,
    int Nb, int C, int H, int W, int OH, int OW, int O, int pad, int relu) {
  __shared__ alignas(32) __bf16 sA[64 * 64];
  __shared__ alignas(32) __bf16 sBt[128 * 64];
  int tid = threadIdx.x, lane = tid & 31, w = tid >> 5;
  int wm = w >> 2, wn = w & 3;
  int K = C * 9;
  int OHW = OH * OW;
  int J = Nb * OHW;
  int mBase = blockIdx.y * 64, nBase = blockIdx.x * 128;
  v8f acc[2][2] = {};
  for (int k0 = 0; k0 < K; k0 += 64) {
    stage_a64x64(wgt, K, mBase, k0, O, K, sA, tid);
    // im2col gather: 8192 elems, 32/thread, batches of 8 to batch VMEM issue
#pragma unroll
    for (int b = 0; b < 4; ++b) {
      float vals[8];
      bool oks[8];
#pragma unroll
      for (int u = 0; u < 8; ++u) {
        int i  = tid + (b * 8 + u) * 256;
        int kk = i >> 7, jl = i & 127;
        int gk = k0 + kk, gj = nBase + jl;
        int c = gk / 9, rem = gk % 9;
        int kh = rem / 3, kw = rem % 3;
        int n = gj / OHW, p = gj % OHW;
        int oy = p / OW, ox = p % OW;
        int iy = oy + kh - pad, ix = ox + kw - pad;
        bool ok = (gj < J) & (gk < K) & (iy >= 0) & (iy < H) & (ix >= 0) & (ix < W);
        long idx = ((long)(n * C + c) * H + iy) * W + ix;
        oks[u] = ok;
        vals[u] = in[ok ? idx : 0];
      }
#pragma unroll
      for (int u = 0; u < 8; ++u) {
        int i  = tid + (b * 8 + u) * 256;
        int kk = i >> 7, jl = i & 127;
        sBt[jl * 64 + kk] = (__bf16)(oks[u] ? vals[u] : 0.f);
      }
    }
    __syncthreads();
    v16bf a00 = load_a_frag(sA, 64, wm * 32,      0,  lane);
    v16bf a01 = load_a_frag(sA, 64, wm * 32,      32, lane);
    v16bf a10 = load_a_frag(sA, 64, wm * 32 + 16, 0,  lane);
    v16bf a11 = load_a_frag(sA, 64, wm * 32 + 16, 32, lane);
    v16bf b00 = load_b_frag(sBt, 64, wn * 32,      0,  lane);
    v16bf b01 = load_b_frag(sBt, 64, wn * 32,      32, lane);
    v16bf b10 = load_b_frag(sBt, 64, wn * 32 + 16, 0,  lane);
    v16bf b11 = load_b_frag(sBt, 64, wn * 32 + 16, 32, lane);
    acc[0][0] = WMMA_BF16(a00, b00, acc[0][0]);
    acc[0][1] = WMMA_BF16(a00, b10, acc[0][1]);
    acc[1][0] = WMMA_BF16(a10, b00, acc[1][0]);
    acc[1][1] = WMMA_BF16(a10, b10, acc[1][1]);
    acc[0][0] = WMMA_BF16(a01, b01, acc[0][0]);
    acc[0][1] = WMMA_BF16(a01, b11, acc[0][1]);
    acc[1][0] = WMMA_BF16(a11, b01, acc[1][0]);
    acc[1][1] = WMMA_BF16(a11, b11, acc[1][1]);
    __syncthreads();
  }
  int l = lane & 15, g = (lane >> 4) & 1;
#pragma unroll
  for (int mi = 0; mi < 2; ++mi)
#pragma unroll
    for (int ni = 0; ni < 2; ++ni)
#pragma unroll
      for (int r = 0; r < 8; ++r) {
        int row = mBase + wm * 32 + mi * 16 + g * 8 + r;  // o
        int col = nBase + wn * 32 + ni * 16 + l;          // pixel j
        if (row < O && col < J) {
          int n = col / OHW, p = col % OHW;
          int oy = p / OW, ox = p % OW;
          int idx = ((n * O + row) * OH + oy) * OW + ox;
          float v = acc[mi][ni][r];
          if (res) v += res[idx];
          if (relu) v = fmaxf(v, 0.f);
          out[idx] = v;
        }
      }
}

// ---------------------------------------------------------------------------
// Fused per-pixel HR2O attention: one block per (h,w) pixel.
//   att = softmax_m( (q . k) / sqrt(C) );  virt = att @ v
// Phase 1: 64x64 GEMM K=512 (WMMA). Phase 2: softmax in LDS.
// Phase 3: 64x512 GEMM K=64 in 128-wide c-chunks (WMMA).
// ---------------------------------------------------------------------------
__global__ __launch_bounds__(256) void hr2o_attention(
    const float* __restrict__ q, const float* __restrict__ kv,
    const float* __restrict__ vv, float* __restrict__ virt, int Cc, int P) {
  int pix = blockIdx.x;
  __shared__ alignas(32) __bf16 sQ[64 * 32];
  __shared__ alignas(32) __bf16 sKt[64 * 32];
  __shared__ alignas(32) float  sAtt[64 * 64];
  __shared__ alignas(32) __bf16 sVt[128 * 64];
  int tid = threadIdx.x, lane = tid & 31, w = tid >> 5;

  // phase 1: att[n,m] = sum_c q[n,c]*k[m,c]
  int wm = w >> 1;
  int wn2 = (w & 1) * 2;
  v8f acc0 = {}, acc1 = {};
  for (int k0 = 0; k0 < Cc; k0 += 32) {
    float fq[8], fk[8];
#pragma unroll
    for (int t = 0; t < 8; ++t) {
      int i = tid + t * 256;
      int r = i >> 5, c = i & 31;
      fq[t] = q [(r * Cc + k0 + c) * P + pix];
      fk[t] = kv[(r * Cc + k0 + c) * P + pix];
    }
#pragma unroll
    for (int t = 0; t < 8; ++t) {
      int i = tid + t * 256;
      sQ[i]  = (__bf16)fq[t];
      sKt[i] = (__bf16)fk[t];
    }
    __syncthreads();
    v16bf a  = load_a_frag(sQ, 32, wm * 16, 0, lane);
    v16bf b0 = load_b_frag(sKt, 32, wn2 * 16,       0, lane);
    v16bf b1 = load_b_frag(sKt, 32, (wn2 + 1) * 16, 0, lane);
    acc0 = WMMA_BF16(a, b0, acc0);
    acc1 = WMMA_BF16(a, b1, acc1);
    __syncthreads();
  }
  float scale = rsqrtf((float)Cc);
  int l = lane & 15, g = (lane >> 4) & 1;
#pragma unroll
  for (int r = 0; r < 8; ++r) {
    int row = wm * 16 + g * 8 + r;
    sAtt[row * 64 + wn2 * 16 + l]       = acc0[r] * scale;
    sAtt[row * 64 + (wn2 + 1) * 16 + l] = acc1[r] * scale;
  }
  __syncthreads();

  // phase 2: softmax over keys m
  if (tid < 64) {
    float mx = -3.402823466e+38f;
    for (int m = 0; m < 64; ++m) mx = fmaxf(mx, sAtt[tid * 64 + m]);
    float s = 0.f;
    for (int m = 0; m < 64; ++m) {
      float e = __expf(sAtt[tid * 64 + m] - mx);
      sAtt[tid * 64 + m] = e;
      s += e;
    }
    float inv = 1.f / s;
    for (int m = 0; m < 64; ++m) sAtt[tid * 64 + m] *= inv;
  }
  __syncthreads();

  // phase 3: virt[n,c] = sum_m att[n,m]*v[m,c], 128-wide c chunks
  for (int c0 = 0; c0 < Cc; c0 += 128) {
#pragma unroll
    for (int b = 0; b < 4; ++b) {
      float fv[8];
#pragma unroll
      for (int u = 0; u < 8; ++u) {
        int i = tid + (b * 8 + u) * 256;
        int cl = i >> 6, m = i & 63;
        fv[u] = vv[(m * Cc + c0 + cl) * P + pix];
      }
#pragma unroll
      for (int u = 0; u < 8; ++u) {
        int i = tid + (b * 8 + u) * 256;
        sVt[i] = (__bf16)fv[u];
      }
    }
    __syncthreads();
    int nm = w >> 1;
    int cb = (w & 1) * 4;
    v16bf a0 = load_a_frag_f32(sAtt, 64, nm * 16, 0,  lane);
    v16bf a1 = load_a_frag_f32(sAtt, 64, nm * 16, 32, lane);
#pragma unroll
    for (int ct = 0; ct < 4; ++ct) {
      v8f acc = {};
      v16bf b0 = load_b_frag(sVt, 64, (cb + ct) * 16, 0,  lane);
      v16bf b1 = load_b_frag(sVt, 64, (cb + ct) * 16, 32, lane);
      acc = WMMA_BF16(a0, b0, acc);
      acc = WMMA_BF16(a1, b1, acc);
#pragma unroll
      for (int r = 0; r < 8; ++r) {
        int n = nm * 16 + g * 8 + r;
        int c = c0 + (cb + ct) * 16 + l;
        virt[(n * Cc + c) * P + pix] = acc[r];
      }
    }
    __syncthreads();
  }
}

// ---------------------------------------------------------------------------
// Elementwise / reduction glue kernels
// ---------------------------------------------------------------------------
__global__ void build_x1(const float* __restrict__ Pm, const float* __restrict__ Qm,
                         float* __restrict__ x1, int total) {
  int idx = blockIdx.x * blockDim.x + threadIdx.x;
  if (idx >= total) return;
  int n = idx / (512 * 256);
  int o = (idx / 256) % 512;
  int hw = idx % 256;
  x1[idx] = fmaxf(Pm[o * 256 + hw] + Qm[o * 64 + n], 0.f);
}

__global__ void maxpool3x3s2(const float* __restrict__ in, float* __restrict__ out,
                             int NC, int H, int W, int OH, int OW) {
  int idx = blockIdx.x * blockDim.x + threadIdx.x;
  int total = NC * OH * OW;
  if (idx >= total) return;
  int nc = idx / (OH * OW), p = idx % (OH * OW);
  int oy = p / OW, ox = p % OW;
  float m = -3.402823466e+38f;
#pragma unroll
  for (int dy = 0; dy < 3; ++dy)
#pragma unroll
    for (int dx = 0; dx < 3; ++dx) {
      int iy = oy * 2 - 1 + dy, ix = ox * 2 - 1 + dx;
      bool ok = (iy >= 0) & (iy < H) & (ix >= 0) & (ix < W);
      float v = in[ok ? ((long)(nc * H + iy) * W + ix) : 0];
      m = fmaxf(m, ok ? v : -3.402823466e+38f);
    }
  out[idx] = m;
}

__global__ void gn_stats(const float* __restrict__ x, float* __restrict__ stats,
                         int per_sample) {
  int n = blockIdx.x, tid = threadIdx.x;
  float s = 0.f, s2 = 0.f;
  const float* xp = x + n * per_sample;
  for (int i = tid; i < per_sample; i += 256) {
    float v = xp[i];
    s += v; s2 += v * v;
  }
  __shared__ float bs[256], bs2[256];
  bs[tid] = s; bs2[tid] = s2;
  __syncthreads();
  for (int o = 128; o > 0; o >>= 1) {
    if (tid < o) { bs[tid] += bs[tid + o]; bs2[tid] += bs2[tid + o]; }
    __syncthreads();
  }
  if (tid == 0) {
    float mu = bs[0] / per_sample;
    float var = bs2[0] / per_sample - mu * mu;
    stats[n * 2]     = mu;
    stats[n * 2 + 1] = rsqrtf(var + 1e-5f);
  }
}

__global__ void gn_norm_relu(float* __restrict__ x, const float* __restrict__ stats,
                             const float* __restrict__ gam, const float* __restrict__ bet,
                             int C, int P, int total) {
  int idx = blockIdx.x * blockDim.x + threadIdx.x;
  if (idx >= total) return;
  int n = idx / (C * P);
  int c = (idx / P) % C;
  float mu = stats[n * 2], rstd = stats[n * 2 + 1];
  float v = (x[idx] - mu) * rstd * gam[c] + bet[c];
  x[idx] = fmaxf(v, 0.f);
}

__global__ void gap_kernel(const float* __restrict__ in, float* __restrict__ out,
                           int NC, int P) {
  int idx = blockIdx.x * blockDim.x + threadIdx.x;
  if (idx >= NC) return;
  float s = 0.f;
  for (int p = 0; p < P; ++p) s += in[idx * P + p];
  out[idx] = s / (float)P;
}

__global__ void concat_kernel(const float* __restrict__ roi, const float* __restrict__ gap,
                              float* __restrict__ cat, int total) {
  int idx = blockIdx.x * blockDim.x + threadIdx.x;
  if (idx >= total) return;
  int n = idx / 1024, j = idx % 1024;
  cat[idx] = (j < 512) ? roi[n * 512 + j] : gap[n * 512 + (j - 512)];
}

// ---------------------------------------------------------------------------
// Host-side orchestration
// ---------------------------------------------------------------------------
extern "C" void kernel_launch(void* const* d_in, const int* in_sizes, int n_in,
                              void* d_out, int out_size, void* d_ws, size_t ws_size,
                              hipStream_t stream) {
  (void)in_sizes; (void)n_in; (void)out_size; (void)ws_size;
  const float* bg    = (const float*)d_in[0];   // [1024,16,16]
  const float* act   = (const float*)d_in[1];   // [64,1024]
  const float* w1    = (const float*)d_in[2];   // [512,2048]
  const float* w2    = (const float*)d_in[3];   // [512,512*9]
  const float* wq    = (const float*)d_in[4];   // [2,512,512*9]
  const float* wk    = (const float*)d_in[5];
  const float* wv    = (const float*)d_in[6];
  const float* wo    = (const float*)d_in[7];
  const float* gam   = (const float*)d_in[8];   // [2,512]
  const float* bet   = (const float*)d_in[9];
  const float* wfc1  = (const float*)d_in[10];  // [512,1024]
  const float* wfc2  = (const float*)d_in[11];  // [60,1024]
  float* out = (float*)d_out;                   // [64,60]
  float* ws  = (float*)d_ws;

  const int Nn = 64, HID = 512, CR = 1024, NCLS = 60;
  const int SZPX = Nn * HID * 49;               // 1,605,632 floats
  float* xpA   = ws;
  float* xpB   = ws + (size_t)SZPX;
  float* qb    = ws + (size_t)2 * SZPX;
  float* kb    = ws + (size_t)3 * SZPX;
  float* vb    = ws + (size_t)4 * SZPX;
  float* virt  = ws + (size_t)5 * SZPX;
  float* x2    = qb;                            // alias: x2 dead before q conv
  float* x1    = ws + (size_t)6 * SZPX;         // 8,388,608 floats
  float* Pbuf  = x1 + (size_t)Nn * HID * 256;   // 512*256
  float* Qbuf  = Pbuf + 512 * 256;              // 512*64
  float* stats = Qbuf + 512 * 64;               // 128
  float* gapb  = stats + 128;                   // 64*512
  float* roib  = gapb + Nn * HID;               // 64*512
  float* catb  = roib + Nn * HID;               // 64*1024

  // --- conv1 split: P = W1[:, :1024] @ bg ; Q = W1[:, 1024:] @ actorT ---
  {
    dim3 grid((256 + 127) / 128, (512 + 63) / 64);
    gemm_bf16_wmma<<<grid, 256, 0, stream>>>(w1, 2 * CR, bg, 256, 0,
                                             Pbuf, 256, 512, 256, CR, 0);
  }
  {
    dim3 grid((64 + 127) / 128, (512 + 63) / 64);
    gemm_bf16_wmma<<<grid, 256, 0, stream>>>(w1 + CR, 2 * CR, act, CR, 1,
                                             Qbuf, 64, 512, 64, CR, 0);
  }
  {
    int total = Nn * HID * 256;
    build_x1<<<(total + 255) / 256, 256, 0, stream>>>(Pbuf, Qbuf, x1, total);
  }

  // --- conv2 3x3 pad=0 : [64,512,16,16] -> [64,512,14,14], relu ---
  {
    int J = Nn * 14 * 14;
    dim3 grid((J + 127) / 128, (HID + 63) / 64);
    conv3x3_wmma<<<grid, 256, 0, stream>>>(x1, w2, x2, nullptr,
                                           Nn, HID, 16, 16, 14, 14, HID, 0, 1);
  }

  // --- maxpool 3x3 s2 pad=1 : 14x14 -> 7x7 ---
  {
    int total = Nn * HID * 49;
    maxpool3x3s2<<<(total + 255) / 256, 256, 0, stream>>>(x2, xpA, Nn * HID, 14, 14, 7, 7);
  }

  // --- HR2O layers ---
  const int WSZ = HID * HID * 9;  // per-layer conv weight elements
  for (int d = 0; d < 2; ++d) {
    float* xin  = (d == 0) ? xpA : xpB;
    float* xout = (d == 0) ? xpB : xpA;
    int J = Nn * 49;
    dim3 cgrid((J + 127) / 128, (HID + 63) / 64);
    conv3x3_wmma<<<cgrid, 256, 0, stream>>>(xin, wq + (size_t)d * WSZ, qb, nullptr,
                                            Nn, HID, 7, 7, 7, 7, HID, 1, 0);
    conv3x3_wmma<<<cgrid, 256, 0, stream>>>(xin, wk + (size_t)d * WSZ, kb, nullptr,
                                            Nn, HID, 7, 7, 7, 7, HID, 1, 0);
    conv3x3_wmma<<<cgrid, 256, 0, stream>>>(xin, wv + (size_t)d * WSZ, vb, nullptr,
                                            Nn, HID, 7, 7, 7, 7, HID, 1, 0);
    hr2o_attention<<<49, 256, 0, stream>>>(qb, kb, vb, virt, HID, 49);
    gn_stats<<<Nn, 256, 0, stream>>>(virt, stats, HID * 49);
    {
      int total = Nn * HID * 49;
      gn_norm_relu<<<(total + 255) / 256, 256, 0, stream>>>(
          virt, stats, gam + d * HID, bet + d * HID, HID, 49, total);
    }
    conv3x3_wmma<<<cgrid, 256, 0, stream>>>(virt, wo + (size_t)d * WSZ, xout, xin,
                                            Nn, HID, 7, 7, 7, 7, HID, 1, 0);
  }

  // --- head: GAP, fc1(relu), concat, fc2 ---
  gap_kernel<<<(Nn * HID + 255) / 256, 256, 0, stream>>>(xpA, gapb, Nn * HID, 49);
  {
    dim3 grid((HID + 127) / 128, (Nn + 63) / 64);
    gemm_bf16_wmma<<<grid, 256, 0, stream>>>(act, CR, wfc1, CR, 1,
                                             roib, HID, Nn, HID, CR, 1);
  }
  {
    int total = Nn * 1024;
    concat_kernel<<<(total + 255) / 256, 256, 0, stream>>>(roib, gapb, catb, total);
  }
  {
    dim3 grid((NCLS + 127) / 128, (Nn + 63) / 64);
    gemm_bf16_wmma<<<grid, 256, 0, stream>>>(catb, 1024, wfc2, 1024, 1,
                                             out, NCLS, Nn, NCLS, 1024, 0);
  }
}
